// TGgBlock_19868518711706
// MI455X (gfx1250) — matrix operand order repaired and checked by reading, not compiled, not run on hardware
//
#include <hip/hip_runtime.h>

// ---- problem constants (match reference) ----
#define B_   4
#define N_   64
#define T_   128
#define F_   16
#define H_   64
#define EG_  16
#define Eg_  8
#define K_   3
#define PAD_ 4            // max (K-1)/2 * dil = 4 -> temporal halo rows

typedef __bf16 bf16;
typedef bf16  v16bf __attribute__((ext_vector_type(16)));
typedef float v8f   __attribute__((ext_vector_type(8)));

// f32x2 -> packed bf16x2 via compiler-native __bf16 conversion (RNE)
__device__ __forceinline__ unsigned pk2(float a, float b) {
    union { bf16 h[2]; unsigned u; } t;
    t.h[0] = (bf16)a;
    t.h[1] = (bf16)b;
    return t.u;
}
__device__ __forceinline__ unsigned short f2bfu(float f) {
    union { bf16 h; unsigned short u; } t;
    t.h = (bf16)f;
    return t.u;
}
__device__ __forceinline__ bf16 u2bf(unsigned short u) {
    union { unsigned short u; bf16 h; } t;
    t.u = u;
    return t.h;
}
__device__ __forceinline__ float elu(float x) {
    return x > 0.f ? x : (__expf(x) - 1.f);
}

__device__ __forceinline__ v8f wmma_bf16(v16bf a, v16bf b, v8f c) {
    return __builtin_amdgcn_wmma_f32_16x16x32_bf16(
        false, a, false, b, (short)0, c, false, false);
}

// A fragment: row-major bf16 row; elements 0..7 = K[k0+half*8 .. +7],
// elements 8..15 = K[k0+16+half*8 .. +7]  -> two 16B LDS loads.
__device__ __forceinline__ v16bf frag_A(const unsigned short* row, int k0, int half) {
    union { v16bf v; uint4 q[2]; } f;
    f.q[0] = *(const uint4*)(row + k0 + half * 8);
    f.q[1] = *(const uint4*)(row + k0 + 16 + half * 8);
    return f.v;
}
// B fragment from an LDS tile stored TRANSPOSED as [N][K] bf16:
// elements 0..7 = K[koff .. koff+7], 8..15 = K[koff+16 .. koff+23] of column n -> two 16B runs.
__device__ __forceinline__ v16bf frag_B(const unsigned short* rowN, int koff) {
    union { v16bf v; uint4 q[2]; } f;
    f.q[0] = *(const uint4*)(rowN + koff);
    f.q[1] = *(const uint4*)(rowN + koff + 16);
    return f.v;
}

// =====================================================================
// Kernel 1: dilated temporal conv, per-feature-instance weights + ELU
// one block per (bn,f); 128 threads = 4 waves; each wave owns M-tiles
// {wave, wave+4}; B fragments shared across both tiles. Zero halo rows
// in LDS replace per-fragment boundary predication.
// =====================================================================
__global__ __launch_bounds__(128)
void tconv_kernel(const float* __restrict__ x, const float* __restrict__ W,
                  const float* __restrict__ bias, float* __restrict__ y, int dil)
{
    __shared__ __align__(16) unsigned short a_t[T_ + 2 * PAD_][H_]; // 17 KB, halo rows zero
    __shared__ __align__(16) unsigned short w_t[K_][H_][H_];        // 24 KB, [k][g][h_in]
    __shared__ float b_s[H_];

    const int bn   = blockIdx.x >> 4;
    const int f    = blockIdx.x & 15;
    const int tid  = threadIdx.x;
    const int lane = tid & 31;
    const int wave = tid >> 5;

    // zero halo rows (8 rows x 64)
    for (int q = tid; q < 2 * PAD_ * (H_ / 4); q += 128) {
        int r  = q >> 4;               // 0..7
        int h4 = (q & 15) * 4;
        int row = (r < PAD_) ? r : (T_ + PAD_ + r - PAD_);
        *(uint2*)&a_t[row][h4] = make_uint2(0u, 0u);
    }
    // activations: contiguous float4 -> packed bf16 uint2 stores
    for (int q = tid; q < T_ * H_ / 4; q += 128) {
        int idx = q * 4;
        int t = idx >> 6, h4 = idx & 63;
        float4 v = *(const float4*)(x + ((size_t)(bn * T_ + t) * F_ + f) * H_ + h4);
        *(uint2*)&a_t[PAD_ + t][h4] = make_uint2(pk2(v.x, v.y), pk2(v.z, v.w));
    }
    // weights transposed: w_t[k][g][h_in] = W[k,f,h_in,g]
    for (int q = tid; q < K_ * H_ * (H_ / 4); q += 128) {
        int k   = q >> 10;
        int rem = q & 1023;
        int g   = rem & 63;
        int h4  = (rem >> 6) * 4;
        const float* wp = W + ((size_t)(k * F_ + f) * H_) * H_ + g;
        float v0 = wp[(h4 + 0) * H_], v1 = wp[(h4 + 1) * H_];
        float v2 = wp[(h4 + 2) * H_], v3 = wp[(h4 + 3) * H_];
        *(uint2*)&w_t[k][g][h4] = make_uint2(pk2(v0, v1), pk2(v2, v3));
    }
    if (tid < H_) b_s[tid] = bias[f * H_ + tid];
    __syncthreads();

    const int mrow = lane & 15;
    const int half = lane >> 4;
    const int ncol = lane & 15;

    v8f acc[2][4];
    #pragma unroll
    for (int nt = 0; nt < 4; ++nt) {
        float bv = b_s[nt * 16 + ncol];
        #pragma unroll
        for (int r = 0; r < 8; ++r) { acc[0][nt][r] = bv; acc[1][nt][r] = bv; }
    }

    #pragma unroll
    for (int k = 0; k < K_; ++k) {
        const int row0 = PAD_ + wave * 16 + mrow + (k - 1) * dil;   // always in-bounds
        #pragma unroll
        for (int kt = 0; kt < 2; ++kt) {
            v16bf A0 = frag_A(&a_t[row0][0],      kt * 32, half);
            v16bf A1 = frag_A(&a_t[row0 + 64][0], kt * 32, half);
            #pragma unroll
            for (int nt = 0; nt < 4; ++nt) {
                v16bf Bf = frag_B(&w_t[k][nt * 16 + ncol][0], kt * 32 + half * 8);
                acc[0][nt] = wmma_bf16(A0, Bf, acc[0][nt]);
                acc[1][nt] = wmma_bf16(A1, Bf, acc[1][nt]);
            }
        }
    }
    #pragma unroll
    for (int mt = 0; mt < 2; ++mt)
        #pragma unroll
        for (int nt = 0; nt < 4; ++nt)
            #pragma unroll
            for (int r = 0; r < 8; ++r) {
                int t = wave * 16 + mt * 64 + r + 8 * half;
                y[((size_t)(bn * T_ + t) * F_ + f) * H_ + nt * 16 + ncol] =
                    elu(acc[mt][nt][r]);
            }
}

// =====================================================================
// Kernel 2 (template): y = ELU( concat(LN(x), emb) @ W + b )
// 256 threads = 8 waves; 128 rows/block; K padded to 96
// =====================================================================
template <int E>
__global__ __launch_bounds__(256)
void lncat_mm_kernel(const float* __restrict__ x, const float* __restrict__ emb,
                     const float* __restrict__ W, const float* __restrict__ bias,
                     const float* __restrict__ gamma, const float* __restrict__ beta,
                     float* __restrict__ y)
{
    __shared__ __align__(16) unsigned short a_t[128][96];   // 24 KB, [row][k]
    __shared__ __align__(16) unsigned short w_t[64][96];    // 12 KB, [g][k]
    __shared__ float b_s[64];

    const int tid  = threadIdx.x;
    const int lane = tid & 31;
    const int wave = tid >> 5;
    const size_t r0 = (size_t)blockIdx.x * 128;

    for (int q = tid; q < 64 * 24; q += 256) {
        int g  = q & 63;
        int r4 = (q >> 6) * 4;
        float v[4];
        #pragma unroll
        for (int j = 0; j < 4; ++j)
            v[j] = (r4 + j < 64 + E) ? W[(r4 + j) * 64 + g] : 0.f;
        *(uint2*)&w_t[g][r4] = make_uint2(pk2(v[0], v[1]), pk2(v[2], v[3]));
    }
    if (tid < 64) b_s[tid] = bias[tid];

    // fused LayerNorm into bf16 A-tile (wave handles its own 16 rows)
    for (int i = 0; i < 16; ++i) {
        int rr = wave * 16 + i;
        size_t gr = (r0 + rr) * 64;
        float v0 = x[gr + lane], v1 = x[gr + 32 + lane];
        float s  = v0 + v1;
        float s2 = v0 * v0 + v1 * v1;
        #pragma unroll
        for (int m = 16; m >= 1; m >>= 1) {
            s  += __shfl_xor(s,  m, 32);
            s2 += __shfl_xor(s2, m, 32);
        }
        float mean = s * (1.f / 64.f);
        float var  = s2 * (1.f / 64.f) - mean * mean;
        float rstd = rsqrtf(var + 1e-5f);
        a_t[rr][lane]      = f2bfu((v0 - mean) * rstd * gamma[lane]      + beta[lane]);
        a_t[rr][32 + lane] = f2bfu((v1 - mean) * rstd * gamma[32 + lane] + beta[32 + lane]);
        if (lane < E)        a_t[rr][64 + lane] = f2bfu(emb[(r0 + rr) * E + lane]);
        else                 a_t[rr][64 + lane] = 0;   // pad cols 64+E..95
    }
    __syncthreads();

    const int mrow = lane & 15;
    const int half = lane >> 4;
    const int ncol = lane & 15;

    v8f acc[4];
    #pragma unroll
    for (int nt = 0; nt < 4; ++nt) {
        float bv = b_s[nt * 16 + ncol];
        #pragma unroll
        for (int r = 0; r < 8; ++r) acc[nt][r] = bv;
    }

    #pragma unroll
    for (int kt = 0; kt < 3; ++kt) {
        v16bf A = frag_A(&a_t[wave * 16 + mrow][0], kt * 32, half);
        #pragma unroll
        for (int nt = 0; nt < 4; ++nt) {
            v16bf Bf = frag_B(&w_t[nt * 16 + ncol][0], kt * 32 + half * 8);
            acc[nt] = wmma_bf16(A, Bf, acc[nt]);
        }
    }

    #pragma unroll
    for (int nt = 0; nt < 4; ++nt)
        #pragma unroll
        for (int r = 0; r < 8; ++r) {
            size_t row = r0 + wave * 16 + r + 8 * half;
            y[row * 64 + nt * 16 + ncol] = elu(acc[nt][r]);
        }
}

// =====================================================================
// Kernel 3: out[b,m,:] = sum_n adj[m,n] * x[b,n,:]   (contraction over N=64)
// 256 threads = 8 waves; block = one batch x 128 columns
// =====================================================================
__global__ __launch_bounds__(256)
void nodeconv_kernel(const float* __restrict__ x, const float* __restrict__ adj,
                     float* __restrict__ y)
{
    __shared__ __align__(16) unsigned short a_t[64][64];    //  8 KB, [m][n]
    __shared__ __align__(16) unsigned short b_t[128][64];   // 16 KB, [c][n]

    const int tid  = threadIdx.x;
    const int lane = tid & 31;
    const int wave = tid >> 5;
    const int b    = blockIdx.x >> 10;
    const size_t c0 = (size_t)(blockIdx.x & 1023) * 128;
    const size_t CS = (size_t)T_ * F_ * H_;   // 131072

    for (int q = tid; q < 64 * 16; q += 256) {
        int idx = q * 4;
        int m = idx >> 6, n4 = idx & 63;
        float4 v = *(const float4*)(adj + m * 64 + n4);
        *(uint2*)&a_t[m][n4] = make_uint2(pk2(v.x, v.y), pk2(v.z, v.w));
    }
    for (int q = tid; q < 128 * 16; q += 256) {
        int c  = q & 127;
        int n4 = (q >> 7) * 4;
        float v0 = x[(size_t)(b * N_ + n4 + 0) * CS + c0 + c];
        float v1 = x[(size_t)(b * N_ + n4 + 1) * CS + c0 + c];
        float v2 = x[(size_t)(b * N_ + n4 + 2) * CS + c0 + c];
        float v3 = x[(size_t)(b * N_ + n4 + 3) * CS + c0 + c];
        *(uint2*)&b_t[c][n4] = make_uint2(pk2(v0, v1), pk2(v2, v3));
    }
    __syncthreads();

    const int mrow = lane & 15;
    const int half = lane >> 4;
    const int ncol = lane & 15;

    v8f acc[4];
    #pragma unroll
    for (int mt = 0; mt < 4; ++mt)
        #pragma unroll
        for (int r = 0; r < 8; ++r) acc[mt][r] = 0.f;

    #pragma unroll
    for (int kt = 0; kt < 2; ++kt) {
        v16bf Bf = frag_B(&b_t[wave * 16 + ncol][0], kt * 32 + half * 8);
        #pragma unroll
        for (int mt = 0; mt < 4; ++mt) {
            v16bf A = frag_A(&a_t[mt * 16 + mrow][0], kt * 32, half);
            acc[mt] = wmma_bf16(A, Bf, acc[mt]);
        }
    }

    #pragma unroll
    for (int mt = 0; mt < 4; ++mt)
        #pragma unroll
        for (int r = 0; r < 8; ++r) {
            int m = mt * 16 + r + 8 * half;
            y[(size_t)(b * N_ + m) * CS + c0 + wave * 16 + ncol] = acc[mt][r];
        }
}

// =====================================================================
// Kernel 5: out[bn,t,g,:] = LN_g( sum_f adj_g[g,f] z[bn,t,f,:] ) + res
// one wave per (bn,t) slice; K=16 zero-padded to 32; LN on the D
// fragment via half-wave shuffles. In-place safe (wave owns its slice).
// =====================================================================
__global__ __launch_bounds__(256)
void featconv_ln_res_kernel(const float* __restrict__ z, const float* __restrict__ adjg,
                            const float* __restrict__ gamma, const float* __restrict__ beta,
                            const float* __restrict__ res, float* __restrict__ y)
{
    __shared__ __align__(16) unsigned short zt[8][64][32];   // 32 KB, [wave][h][f]

    const int tid  = threadIdx.x;
    const int lane = tid & 31;
    const int wave = tid >> 5;
    const size_t pair = (size_t)blockIdx.x * 8 + wave;   // (bn*T + t)
    const float* zp = z + pair * (F_ * H_);

    for (int q = lane; q < 64 * 4; q += 32) {
        int h  = q & 63;
        int f4 = (q >> 6) * 4;
        float v0 = zp[(f4 + 0) * H_ + h], v1 = zp[(f4 + 1) * H_ + h];
        float v2 = zp[(f4 + 2) * H_ + h], v3 = zp[(f4 + 3) * H_ + h];
        *(uint2*)&zt[wave][h][f4]      = make_uint2(pk2(v0, v1), pk2(v2, v3));
        *(uint2*)&zt[wave][h][f4 + 16] = make_uint2(0u, 0u);    // K pad 16..31
    }
    __syncthreads();

    const int mrow = lane & 15;
    const int half = lane >> 4;
    const int ncol = lane & 15;

    // A = adj_g: elements 0..7 = K[half*8 .. +7] (f<16); 8..15 = zero pad
    v16bf A;
    #pragma unroll
    for (int j = 0; j < 8; ++j) A[j] = (bf16)adjg[mrow * 16 + half * 8 + j];
    #pragma unroll
    for (int j = 8; j < 16; ++j) A[j] = u2bf(0);

    v8f acc[4];
    #pragma unroll
    for (int nt = 0; nt < 4; ++nt) {
        #pragma unroll
        for (int r = 0; r < 8; ++r) acc[nt][r] = 0.f;
        v16bf Bf = frag_B(&zt[wave][nt * 16 + ncol][0], half * 8);
        acc[nt] = wmma_bf16(A, Bf, acc[nt]);
    }

    // LayerNorm over H on the D fragment (half-wave butterfly reduce)
    #pragma unroll
    for (int r = 0; r < 8; ++r) {
        float s = 0.f, s2 = 0.f;
        #pragma unroll
        for (int nt = 0; nt < 4; ++nt) { float v = acc[nt][r]; s += v; s2 += v * v; }
        #pragma unroll
        for (int m = 8; m >= 1; m >>= 1) {
            s  += __shfl_xor(s,  m, 16);
            s2 += __shfl_xor(s2, m, 16);
        }
        float mean = s * (1.f / 64.f);
        float var  = s2 * (1.f / 64.f) - mean * mean;
        float rstd = rsqrtf(var + 1e-5f);
        int g = r + 8 * half;
        #pragma unroll
        for (int nt = 0; nt < 4; ++nt) {
            int hcol = nt * 16 + ncol;
            size_t oi = pair * (F_ * H_) + (size_t)g * H_ + hcol;
            y[oi] = (acc[nt][r] - mean) * rstd * gamma[hcol] + beta[hcol] + res[oi];
        }
    }
}

// =====================================================================
extern "C" void kernel_launch(void* const* d_in, const int* in_sizes, int n_in,
                              void* d_out, int out_size, void* d_ws, size_t ws_size,
                              hipStream_t stream)
{
    (void)in_sizes; (void)n_in; (void)out_size; (void)ws_size;

    const float* h       = (const float*)d_in[0];
    // d_in[1] = mask : unused by the reference
    const float* embsG   = (const float*)d_in[2];
    const float* embsg   = (const float*)d_in[3];
    const float* adjG    = (const float*)d_in[4];
    const float* adjg    = (const float*)d_in[5];
    const float* Wt0     = (const float*)d_in[6];
    const float* bt0     = (const float*)d_in[7];
    const float* Wt1     = (const float*)d_in[8];
    const float* bt1     = (const float*)d_in[9];
    const float* Wt2     = (const float*)d_in[10];
    const float* bt2     = (const float*)d_in[11];
    const float* gamma_t = (const float*)d_in[12];
    const float* beta_t  = (const float*)d_in[13];
    const float* W_G     = (const float*)d_in[14];
    const float* b_G     = (const float*)d_in[15];
    const float* gamma_G = (const float*)d_in[16];
    const float* beta_G  = (const float*)d_in[17];
    const float* W_g     = (const float*)d_in[18];
    const float* b_g     = (const float*)d_in[19];
    const float* gamma_g = (const float*)d_in[20];
    const float* beta_g  = (const float*)d_in[21];

    float* out = (float*)d_out;
    float* buf = (float*)d_ws;   // one f32 ping buffer [BN,T,F,H] = 134 MB

    // temporal conv stack (dilations 1,2,4), ping-pong ws <-> out
    tconv_kernel<<<4096, 128, 0, stream>>>(h,   Wt0, bt0, buf, 1);
    tconv_kernel<<<4096, 128, 0, stream>>>(buf, Wt1, bt1, out, 2);
    tconv_kernel<<<4096, 128, 0, stream>>>(out, Wt2, bt2, buf, 4);
    // LN_t + concat(embs_G) @ W_G + ELU
    lncat_mm_kernel<EG_><<<4096, 256, 0, stream>>>(buf, embsG, W_G, b_G,
                                                   gamma_t, beta_t, out);
    // node-level graph conv (adj_G)
    nodeconv_kernel<<<4096, 256, 0, stream>>>(out, adjG, buf);
    // LN_G + concat(embs_g) @ W_g + ELU
    lncat_mm_kernel<Eg_><<<4096, 256, 0, stream>>>(buf, embsg, W_g, b_g,
                                                   gamma_G, beta_G, out);
    // feature-level graph conv (adj_g) + LN_g + residual, in-place on out
    featconv_ln_res_kernel<<<4096, 256, 0, stream>>>(out, adjg, gamma_g, beta_g,
                                                     h, out);
}
